// SelectiveSSMCore_6262062318294
// MI455X (gfx1250) — matrix operand order repaired and checked by reading, not compiled, not run on hardware
//
#include <hip/hip_runtime.h>
#include <math.h>

// ---- problem constants (match reference) ----
#define BB      2
#define LL      2048
#define DM      512      // d_model
#define DS      16       // d_state
#define DTR     32       // dt_rank
#define NOUT    64       // DTR + 2*DS
#define ROWS    (BB*LL)  // 4096
#define NCH     16       // scan chunks
#define CLEN    (LL/NCH) // 128
#define NCHAN   (BB*DM)  // 1024 channels

typedef __attribute__((ext_vector_type(16))) _Float16 v16h;
typedef __attribute__((ext_vector_type(8)))  float    v8f;

// =====================================================================
// Kernel 1: fused front-end.
// One wave per 16-row tile of (B*L) rows:
//   xz(16x64) = x_tile(16x512) @ W_x^T          -- 64 v_wmma (4 N-tiles x 16 K-steps)
//   dt(16x512) = clip(softplus(dt_raw(16x32) @ W_dt^T + b_dt)) -- 32 v_wmma
// Writes: dt_ws[ROWS][DM], Bm_ws[ROWS][DS], Cm_ws[ROWS][DS]
// =====================================================================
__global__ __launch_bounds__(128)
void k_ssm_frontend(const float* __restrict__ x,
                    const float* __restrict__ W_x,
                    const float* __restrict__ W_dt,
                    const float* __restrict__ b_dt,
                    float* __restrict__ dt_ws,
                    float* __restrict__ Bm_ws,
                    float* __restrict__ Cm_ws)
{
    __shared__ float sdt[4][16][DTR];          // per-wave dt_raw re-layout tile

    const int wv      = threadIdx.x >> 5;      // wave in block (0..3)
    const int lane    = threadIdx.x & 31;
    const int rowtile = blockIdx.x * 4 + wv;   // 0..255
    const int m       = lane & 15;             // tile row (A/B lane id)
    const int hi      = lane >> 4;             // lane-half (K-half select)

    v8f acc0 = {}, acc1 = {}, acc2 = {}, acc3 = {};

    // ---------------- GEMM 1: K = 512 in 16 steps of 32 ----------------
    const float* xrow = x + (size_t)(rowtile * 16 + m) * DM;
    #pragma unroll 4
    for (int ks = 0; ks < 16; ++ks) {
        const int k0 = ks * 32 + hi * 16;      // A-layout: contiguous 16 K per lane-half
        v16h a;
        #pragma unroll
        for (int e = 0; e < 8; ++e) {
            a[e]     = (_Float16)xrow[k0 + e];
            a[e + 8] = (_Float16)xrow[k0 + 8 + e];
        }
        const int kb = ks * 32 + hi * 16;      // B-layout: contiguous 16 K per lane-half
        #pragma unroll
        for (int nt = 0; nt < 4; ++nt) {
            const float* wr = W_x + (size_t)(nt * 16 + m) * DM + kb;
            v16h bm;
            #pragma unroll
            for (int e = 0; e < 16; ++e) bm[e] = (_Float16)wr[e];
            switch (nt) {
            case 0: acc0 = __builtin_amdgcn_wmma_f32_16x16x32_f16(false, a, false, bm, (short)0, acc0, false, false); break;
            case 1: acc1 = __builtin_amdgcn_wmma_f32_16x16x32_f16(false, a, false, bm, (short)0, acc1, false, false); break;
            case 2: acc2 = __builtin_amdgcn_wmma_f32_16x16x32_f16(false, a, false, bm, (short)0, acc2, false, false); break;
            default: acc3 = __builtin_amdgcn_wmma_f32_16x16x32_f16(false, a, false, bm, (short)0, acc3, false, false); break;
            }
        }
    }

    // ---- scatter results: C/D layout is [vgpr r] -> row r + 8*hi, lane&15 -> col ----
    #pragma unroll
    for (int r = 0; r < 8; ++r) {
        const int mr  = r + hi * 8;
        const int row = rowtile * 16 + mr;
        // dt_raw (n = 0..31) -> LDS for re-layout
        sdt[wv][mr][m]      = acc0[r];
        sdt[wv][mr][m + 16] = acc1[r];
        // B_mat (n = 32..47), C_mat (n = 48..63) -> workspace
        Bm_ws[(size_t)row * DS + m] = acc2[r];
        Cm_ws[(size_t)row * DS + m] = acc3[r];
    }
    __syncthreads();

    // ---------------- GEMM 2: dt = softplus(dt_raw @ W_dt^T + b_dt) ----------------
    // A (16x32 f16) from LDS in WMMA A-layout
    v16h a2;
    #pragma unroll
    for (int e = 0; e < 16; ++e) {
        const int k = hi * 16 + e;
        a2[e] = (_Float16)sdt[wv][m][k];
    }

    for (int nt = 0; nt < DM / 16; ++nt) {     // 32 N-tiles of 16 model-dims
        const int d = nt * 16 + m;             // this lane's output column (model dim)
        const float* wr = W_dt + (size_t)d * DTR + hi * 16;
        v16h bm;
        #pragma unroll
        for (int e = 0; e < 16; ++e) bm[e] = (_Float16)wr[e];
        v8f c = {};
        c = __builtin_amdgcn_wmma_f32_16x16x32_f16(false, a2, false, bm, (short)0, c, false, false);
        const float bias = b_dt[d];
        #pragma unroll
        for (int r = 0; r < 8; ++r) {
            const int row = rowtile * 16 + r + hi * 8;
            float v  = c[r] + bias;
            float sp = (v > 20.0f) ? v : log1pf(__expf(v));      // softplus
            sp = fminf(fmaxf(sp, 1e-4f), 10.0f);                 // clip
            dt_ws[(size_t)row * DM + d] = sp;
        }
    }
}

// =====================================================================
// Kernel 2 (scan phase A): per-chunk summaries (prod A_bar, h_end).
// One wave handles 2 channels (16 states each) for one chunk of 128 steps.
// =====================================================================
__global__ __launch_bounds__(128)
void k_scan_partial(const float* __restrict__ x,
                    const float* __restrict__ A_log,
                    const float* __restrict__ dt_ws,
                    const float* __restrict__ Bm_ws,
                    float* __restrict__ Aprod,
                    float* __restrict__ Hend)
{
    const int lane  = threadIdx.x & 31;
    const int wid   = blockIdx.x * 4 + (threadIdx.x >> 5);  // 0..8191
    const int pair  = wid >> 4;                             // 0..511
    const int chunk = wid & 15;                             // 0..15
    const int c     = pair * 2 + (lane >> 4);               // channel 0..1023
    const int b     = c >> 9;
    const int d     = c & (DM - 1);
    const int s     = lane & 15;

    const float A_ds = fminf(-__expf(A_log[d * DS + s]), -1e-4f);

    float a_acc = 1.0f, h = 0.0f;
    const int lbase = chunk * CLEN;
    for (int i = 0; i < CLEN; ++i) {
        const size_t row = (size_t)b * LL + (lbase + i);
        const float dt_v = dt_ws[row * DM + d];
        const float x_v  = x[row * DM + d];
        const float Bm   = Bm_ws[row * DS + s];
        const float a_l  = fminf(__expf(dt_v * A_ds), 1.0f);
        h     = __builtin_fmaf(a_l, h, dt_v * Bm * x_v);
        a_acc *= a_l;
    }
    const size_t idx = ((size_t)c * NCH + chunk) * DS + s;
    Aprod[idx] = a_acc;
    Hend[idx]  = h;
}

// =====================================================================
// Kernel 3 (scan phase B): serial combine over the 16 chunk summaries.
// One thread per (channel, state): emits the incoming state of each chunk.
// =====================================================================
__global__ __launch_bounds__(256)
void k_scan_combine(const float* __restrict__ Aprod,
                    const float* __restrict__ Hend,
                    float* __restrict__ Hinit)
{
    const int tid = blockIdx.x * blockDim.x + threadIdx.x;  // 0..16383
    const int c = tid >> 4;
    const int s = tid & 15;
    float carry = 0.0f;
    #pragma unroll
    for (int ch = 0; ch < NCH; ++ch) {
        const size_t idx = ((size_t)c * NCH + ch) * DS + s;
        Hinit[idx] = carry;
        carry = __builtin_fmaf(Aprod[idx], carry, Hend[idx]);
    }
}

// =====================================================================
// Kernel 4 (scan phase C): replay each chunk from corrected init state,
// contract over states with C_mat (width-16 shuffle reduce), emit y.
// =====================================================================
__global__ __launch_bounds__(128)
void k_scan_final(const float* __restrict__ x,
                  const float* __restrict__ A_log,
                  const float* __restrict__ Dg,
                  const float* __restrict__ dt_ws,
                  const float* __restrict__ Bm_ws,
                  const float* __restrict__ Cm_ws,
                  const float* __restrict__ Hinit,
                  float* __restrict__ y)
{
    const int lane  = threadIdx.x & 31;
    const int wid   = blockIdx.x * 4 + (threadIdx.x >> 5);
    const int pair  = wid >> 4;
    const int chunk = wid & 15;
    const int c     = pair * 2 + (lane >> 4);
    const int b     = c >> 9;
    const int d     = c & (DM - 1);
    const int s     = lane & 15;

    const float A_ds = fminf(-__expf(A_log[d * DS + s]), -1e-4f);
    const float D_d  = Dg[d];

    float h = Hinit[((size_t)c * NCH + chunk) * DS + s];
    const int lbase = chunk * CLEN;
    for (int i = 0; i < CLEN; ++i) {
        const size_t row = (size_t)b * LL + (lbase + i);
        if (i + 4 < CLEN) {
            __builtin_prefetch(&dt_ws[(row + 4) * DM + d], 0, 0);
            __builtin_prefetch(&x[(row + 4) * DM + d], 0, 0);
        }
        const float dt_v = dt_ws[row * DM + d];
        const float x_v  = x[row * DM + d];
        const float Bm   = Bm_ws[row * DS + s];
        const float Cm   = Cm_ws[row * DS + s];
        const float a_l  = fminf(__expf(dt_v * A_ds), 1.0f);
        h = __builtin_fmaf(a_l, h, dt_v * Bm * x_v);

        float ysum = h * Cm;                       // reduce 16 states per half-wave
        #pragma unroll
        for (int off = 8; off > 0; off >>= 1)
            ysum += __shfl_xor(ysum, off, 16);
        if (s == 0)
            y[row * DM + d] = ysum + x_v * D_d;
    }
}

// =====================================================================
extern "C" void kernel_launch(void* const* d_in, const int* in_sizes, int n_in,
                              void* d_out, int out_size, void* d_ws, size_t ws_size,
                              hipStream_t stream)
{
    (void)in_sizes; (void)n_in; (void)out_size; (void)ws_size;
    const float* x     = (const float*)d_in[0];   // (B,L,DM)
    const float* W_x   = (const float*)d_in[1];   // (64, DM)
    const float* W_dt  = (const float*)d_in[2];   // (DM, 32)
    const float* b_dt  = (const float*)d_in[3];   // (DM,)
    const float* A_log = (const float*)d_in[4];   // (DM, DS)
    const float* Dg    = (const float*)d_in[5];   // (DM,)
    float* y = (float*)d_out;

    // workspace layout (floats)
    float* dt_ws = (float*)d_ws;                              // ROWS*DM   = 2,097,152
    float* Bm_ws = dt_ws + (size_t)ROWS * DM;                 // ROWS*DS   =    65,536
    float* Cm_ws = Bm_ws + (size_t)ROWS * DS;                 // ROWS*DS
    float* Aprod = Cm_ws + (size_t)ROWS * DS;                 // NCHAN*NCH*DS = 262,144
    float* Hend  = Aprod + (size_t)NCHAN * NCH * DS;
    float* Hinit = Hend  + (size_t)NCHAN * NCH * DS;

    // 1) fused projections + dt GEMM (256 waves, 4 per block)
    k_ssm_frontend<<<64, 128, 0, stream>>>(x, W_x, W_dt, b_dt, dt_ws, Bm_ws, Cm_ws);
    // 2) per-chunk scan summaries (8192 wave tasks)
    k_scan_partial<<<2048, 128, 0, stream>>>(x, A_log, dt_ws, Bm_ws, Aprod, Hend);
    // 3) combine chunk summaries (16K threads)
    k_scan_combine<<<64, 256, 0, stream>>>(Aprod, Hend, Hinit);
    // 4) replay + output
    k_scan_final<<<2048, 128, 0, stream>>>(x, A_log, Dg, dt_ws, Bm_ws, Cm_ws, Hinit, y);
}
